// ViTMAEEmbeddings_36275293782026
// MI455X (gfx1250) — matrix operand order repaired and checked by reading, not compiled, not run on hardware
//
#include <hip/hip_runtime.h>
#include <hip/hip_bf16.h>

// ---------------------------------------------------------------------------
// ViT-MAE embeddings for MI455X (gfx1250, wave32, WMMA).
// Outputs (flat, concatenated): embeddings(256,50,768) | mask(256,196) |
// ids_restore(256,196 as float).
//
// Pipeline: mask_sort -> pack_w(bf16) -> pack_a(gather kept patches, bf16)
//           -> cls rows -> WMMA GEMM (bf16, f32 acc, double-buffered).
// ---------------------------------------------------------------------------

#define BATCH     256
#define HIDDEN    768
#define KDIM      768          // 3*16*16
#define GRID_P    14
#define NPATCH    196
#define LEN_KEEP  49
#define MPAD      64           // kept rows padded to 64 (2 x 32 M-tiles)
#define OUT_ROWS  50           // 1 cls + 49 kept
#define IMG_CH_STRIDE  (224*224)     // 50176 floats
#define IMG_B_STRIDE   (3*224*224)   // 150528 floats

typedef __attribute__((ext_vector_type(16))) __bf16 v16bf;
typedef __attribute__((ext_vector_type(8)))  __bf16 v8bf;
typedef __attribute__((ext_vector_type(8)))  float  v8f;

// Workspace layout (bytes):
//   [0)        ids_keep : int[256*49]      =    50,176
//   [WS_WOFF)  w_bf16   : bf16[768*768]    = 1,179,648
//   [WS_AOFF)  a_bf16   : bf16[256*64*768] = 25,165,824
#define WS_WOFF   50176
#define WS_AOFF   (WS_WOFF + 2*KDIM*HIDDEN)

// ---------------------------------------------------------------------------
// Kernel 1: per-batch stable argsort of noise (rank counting) -> mask,
// ids_restore, and the ids_keep gather table.
// ---------------------------------------------------------------------------
__global__ __launch_bounds__(256) void mask_sort_kernel(
    const float* __restrict__ noise,
    float* __restrict__ mask_out,
    float* __restrict__ idsres_out,
    int*   __restrict__ ids_keep)
{
    __shared__ float s[NPATCH];
    const int b = blockIdx.x;
    const int t = threadIdx.x;
    if (t < NPATCH) s[t] = noise[b * NPATCH + t];
    __syncthreads();
    if (t < NPATCH) {
        const float v = s[t];
        int rank = 0;
        #pragma unroll 4
        for (int j = 0; j < NPATCH; ++j) {
            float u = s[j];
            rank += (u < v) || (u == v && j < t);   // stable tie-break
        }
        idsres_out[b * NPATCH + t] = (float)rank;
        mask_out[b * NPATCH + t]   = (rank < LEN_KEEP) ? 0.0f : 1.0f;
        if (rank < LEN_KEEP) ids_keep[b * LEN_KEEP + rank] = t;
    }
}

// ---------------------------------------------------------------------------
// Kernel 2: W (768x768 f32) -> bf16, once. One thread = 8 consecutive k.
// ---------------------------------------------------------------------------
__global__ __launch_bounds__(256) void pack_w_kernel(
    const float* __restrict__ w, __bf16* __restrict__ wbf)
{
    const int i = blockIdx.x * 256 + threadIdx.x;     // 768*768/8 = 73728
    const float* p = w + (size_t)i * 8;
    float4 x0 = *(const float4*)(p);
    float4 x1 = *(const float4*)(p + 4);
    v8bf o;
    o[0] = (__bf16)x0.x; o[1] = (__bf16)x0.y; o[2] = (__bf16)x0.z; o[3] = (__bf16)x0.w;
    o[4] = (__bf16)x1.x; o[5] = (__bf16)x1.y; o[6] = (__bf16)x1.z; o[7] = (__bf16)x1.w;
    *(v8bf*)(wbf + (size_t)i * 8) = o;
}

// ---------------------------------------------------------------------------
// Kernel 3: gather the 49 kept patches per image, convert to bf16, store
// dense row-major [b][64][768] (rows 49..63 zero). One thread = 8 k values.
// ---------------------------------------------------------------------------
__global__ __launch_bounds__(256) void pack_a_kernel(
    const float* __restrict__ pix,
    const int*   __restrict__ ids_keep,
    __bf16* __restrict__ abf)
{
    const int i  = blockIdx.x * 256 + threadIdx.x;    // 256*64*96 chunks
    const int kc = i % (KDIM / 8);
    const int row = (i / (KDIM / 8)) & (MPAD - 1);
    const int b   = i / ((KDIM / 8) * MPAD);

    __bf16* dst = abf + ((size_t)(b * MPAD + row) * KDIM + kc * 8);
    if (row < LEN_KEEP) {
        const int patch = ids_keep[b * LEN_KEEP + row];
        const int gh = patch / GRID_P, gw = patch % GRID_P;
        const int k  = kc * 8;
        const int c  = k >> 8, ph = (k >> 4) & 15, pw = k & 15;  // pw in {0,8}
        const float* p = pix + (size_t)b * IMG_B_STRIDE + c * IMG_CH_STRIDE
                             + (gh * 16 + ph) * 224 + gw * 16 + pw;
        float4 x0 = *(const float4*)(p);
        float4 x1 = *(const float4*)(p + 4);
        v8bf o;
        o[0] = (__bf16)x0.x; o[1] = (__bf16)x0.y; o[2] = (__bf16)x0.z; o[3] = (__bf16)x0.w;
        o[4] = (__bf16)x1.x; o[5] = (__bf16)x1.y; o[6] = (__bf16)x1.z; o[7] = (__bf16)x1.w;
        *(v8bf*)dst = o;
    } else {
        v8bf z = {};
        *(v8bf*)dst = z;
    }
}

// ---------------------------------------------------------------------------
// Kernel 4: CLS row: out[b,0,:] = cls_token + pos_embed[0,:]
// ---------------------------------------------------------------------------
__global__ __launch_bounds__(256) void cls_kernel(
    const float* __restrict__ cls,
    const float* __restrict__ pos,
    float* __restrict__ out)
{
    int i = blockIdx.x * 256 + threadIdx.x;     // 256*768 total
    if (i < BATCH * HIDDEN) {
        int b = i / HIDDEN, n = i % HIDDEN;
        out[(size_t)b * (OUT_ROWS * HIDDEN) + n] = cls[n] + pos[n];
    }
}

// ---------------------------------------------------------------------------
// A-fragment loader: 16-bit A 16x32, contiguous k order (matches B).
//   lanes 0-15:  elems 0..15 = K=k0..k0+15
//   lanes 16-31: elems 0..15 = K=k0+16..k0+31
// (per-lane base already includes laneHi*16; here we fetch +0 and +8)
// ---------------------------------------------------------------------------
static __device__ __forceinline__ v16bf load_a_frag(const __bf16* p) {
    v8bf lo = *(const v8bf*)(p);
    v8bf hi = *(const v8bf*)(p + 8);
    return __builtin_shufflevector(lo, hi,
        0, 1, 2, 3, 4, 5, 6, 7, 8, 9, 10, 11, 12, 13, 14, 15);
}

// ---------------------------------------------------------------------------
// Kernel 5: WMMA GEMM. One wave -> 32(M) x 64(N) tile, K=768 in 24 steps,
// 8 x v_wmma_f32_16x16x32_bf16 per step, double-buffered operand fetch
// (next k-chunk loads issued before current WMMAs) -> loads overlap compute.
// 12 b128 loads per 8 WMMAs.
// ---------------------------------------------------------------------------
__global__ __launch_bounds__(256) void patch_gemm_kernel(
    const __bf16* __restrict__ abf,     // (256, 64, 768) bf16
    const __bf16* __restrict__ wbf,     // (768, 768) bf16
    const float*  __restrict__ bias,    // (768)
    const float*  __restrict__ pos,     // (197, 768)
    const int*    __restrict__ ids_keep,// (256, 49)
    float* __restrict__ out)
{
    const int wave = threadIdx.x >> 5;
    const int lane = threadIdx.x & 31;

    // wave-tile id: 12 n64-tiles x 2 m32-tiles x 256 batches = 6144 waves
    const int t     = blockIdx.x * 8 + wave;
    const int n64   = (t % 12) * 64;
    const int m32   = ((t / 12) & 1) * 32;
    const int b     = t / 24;

    const int laneM  = lane & 15;
    const int laneHi = lane >> 4;

    const __bf16* aptr0 = abf + ((size_t)(b * MPAD + m32 + laneM) * KDIM) + laneHi * 16;
    const __bf16* aptr1 = aptr0 + 16 * KDIM;

    // B: column n = lane%16; lanes 0-15 K=k0..k0+15, lanes 16-31 K=k0+16..k0+31
    const __bf16* bptr0 = wbf + (size_t)(n64 +  0 + laneM) * KDIM + laneHi * 16;
    const __bf16* bptr1 = wbf + (size_t)(n64 + 16 + laneM) * KDIM + laneHi * 16;
    const __bf16* bptr2 = wbf + (size_t)(n64 + 32 + laneM) * KDIM + laneHi * 16;
    const __bf16* bptr3 = wbf + (size_t)(n64 + 48 + laneM) * KDIM + laneHi * 16;

    v8f acc[8] = {};

    // ---- prologue: fetch k-chunk 0 ----
    v16bf A0 = load_a_frag(aptr0);
    v16bf A1 = load_a_frag(aptr1);
    v16bf B0 = *(const v16bf*)(bptr0);
    v16bf B1 = *(const v16bf*)(bptr1);
    v16bf B2 = *(const v16bf*)(bptr2);
    v16bf B3 = *(const v16bf*)(bptr3);

    #pragma unroll 2
    for (int k0 = 0; k0 < KDIM; k0 += 32) {
        // issue next-chunk loads first so the 8 WMMAs hide their latency
        const int kn = (k0 + 32 < KDIM) ? (k0 + 32) : 0;   // last iter: dummy reload
        v16bf nA0 = load_a_frag(aptr0 + kn);
        v16bf nA1 = load_a_frag(aptr1 + kn);
        v16bf nB0 = *(const v16bf*)(bptr0 + kn);
        v16bf nB1 = *(const v16bf*)(bptr1 + kn);
        v16bf nB2 = *(const v16bf*)(bptr2 + kn);
        v16bf nB3 = *(const v16bf*)(bptr3 + kn);

        // speculative prefetch two chunks ahead on the A streams (branchless;
        // OOB speculative prefetches are dropped by hardware)
        __builtin_prefetch(aptr0 + k0 + 64, 0, 3);
        __builtin_prefetch(aptr1 + k0 + 64, 0, 3);

        acc[0] = __builtin_amdgcn_wmma_f32_16x16x32_bf16(false, A0, false, B0,
                                                         (short)0, acc[0], false, false);
        acc[1] = __builtin_amdgcn_wmma_f32_16x16x32_bf16(false, A0, false, B1,
                                                         (short)0, acc[1], false, false);
        acc[2] = __builtin_amdgcn_wmma_f32_16x16x32_bf16(false, A0, false, B2,
                                                         (short)0, acc[2], false, false);
        acc[3] = __builtin_amdgcn_wmma_f32_16x16x32_bf16(false, A0, false, B3,
                                                         (short)0, acc[3], false, false);
        acc[4] = __builtin_amdgcn_wmma_f32_16x16x32_bf16(false, A1, false, B0,
                                                         (short)0, acc[4], false, false);
        acc[5] = __builtin_amdgcn_wmma_f32_16x16x32_bf16(false, A1, false, B1,
                                                         (short)0, acc[5], false, false);
        acc[6] = __builtin_amdgcn_wmma_f32_16x16x32_bf16(false, A1, false, B2,
                                                         (short)0, acc[6], false, false);
        acc[7] = __builtin_amdgcn_wmma_f32_16x16x32_bf16(false, A1, false, B3,
                                                         (short)0, acc[7], false, false);

        A0 = nA0; A1 = nA1;
        B0 = nB0; B1 = nB1; B2 = nB2; B3 = nB3;
    }

    // ----- epilogue: +bias, +pos_embed(gathered), store rows 1..49 -----
    // D layout: lanes 0-15: N=lane, VGPR r -> M=r; lanes 16-31: N=lane-16, M=r+8.
    #pragma unroll
    for (int mh = 0; mh < 2; ++mh) {
        #pragma unroll
        for (int r = 0; r < 8; ++r) {
            const int m    = r + laneHi * 8;
            const int mrow = m32 + mh * 16 + m;
            if (mrow < LEN_KEEP) {
                const int patch = ids_keep[b * LEN_KEEP + mrow];
                const float* posrow = pos + (size_t)(1 + patch) * HIDDEN;
                float* orow = out + (size_t)b * (OUT_ROWS * HIDDEN)
                                  + (size_t)(1 + mrow) * HIDDEN;
                #pragma unroll
                for (int nt = 0; nt < 4; ++nt) {
                    const int n = n64 + nt * 16 + laneM;
                    orow[n] = acc[mh * 4 + nt][r] + bias[n] + posrow[n];
                }
            }
        }
    }
}

// ---------------------------------------------------------------------------
extern "C" void kernel_launch(void* const* d_in, const int* in_sizes, int n_in,
                              void* d_out, int out_size, void* d_ws, size_t ws_size,
                              hipStream_t stream)
{
    const float* pix   = (const float*)d_in[0];  // (256,3,224,224)
    const float* noise = (const float*)d_in[1];  // (256,196)
    const float* w     = (const float*)d_in[2];  // (768,3,16,16) -> (768,768)
    const float* bias  = (const float*)d_in[3];  // (768)
    const float* cls   = (const float*)d_in[4];  // (1,1,768)
    const float* pos   = (const float*)d_in[5];  // (1,197,768)

    float* out      = (float*)d_out;
    float* mask_out = out + (size_t)BATCH * OUT_ROWS * HIDDEN;   // 9,830,400
    float* ids_out  = mask_out + (size_t)BATCH * NPATCH;         // +50,176

    char*   ws       = (char*)d_ws;
    int*    ids_keep = (int*)ws;                                 // 256*49 ints
    __bf16* wbf      = (__bf16*)(ws + WS_WOFF);                  // 768*768 bf16
    __bf16* abf      = (__bf16*)(ws + WS_AOFF);                  // 256*64*768 bf16

    // 1) sort/mask/ids (fills ids_keep used by the gather)
    mask_sort_kernel<<<BATCH, 256, 0, stream>>>(noise, mask_out, ids_out, ids_keep);

    // 2) W -> bf16 (768*768/8 = 73728 threads)
    pack_w_kernel<<<(KDIM * HIDDEN / 8) / 256, 256, 0, stream>>>(w, wbf);

    // 3) gather kept patches -> bf16 A (256*64*96 = 1,572,864 threads)
    pack_a_kernel<<<(BATCH * MPAD * (KDIM / 8)) / 256, 256, 0, stream>>>(pix, ids_keep, abf);

    // 4) CLS rows
    cls_kernel<<<(BATCH * HIDDEN + 255) / 256, 256, 0, stream>>>(cls, pos, out);

    // 5) WMMA GEMM: 256 b * 2 m32 * 12 n64 = 6144 wave-tiles / 8 per block
    patch_gemm_kernel<<<6144 / 8, 256, 0, stream>>>(abf, wbf, bias, pos, ids_keep, out);
}
